// CrossAttention_63213328662955
// MI455X (gfx1250) — matrix operand order repaired and checked by reading, not compiled, not run on hardware
//
#include <hip/hip_runtime.h>
#include <stdint.h>

// ---------------------------------------------------------------------------
// Cross-attention, MI455X (gfx1250, wave32, WMMA bf16 16x16x32)
// B=8, CQ=256, CKV=512, H=W=64 -> N=4096, DQK=32 (== one WMMA K-depth)
// Flash attention stage uses double-buffered GLOBAL_LOAD_ASYNC_TO_LDS_B128
// staging of K/V chunks shared by the 4 waves of a block (ASYNCcnt pipeline).
// ---------------------------------------------------------------------------

#define BB   8
#define CQn  256
#define CKVn 512
#define NN   4096
#define DQKn 32
#define HKV  32
#define WKV  32
#define MCH  32          // m-chunk per flash iteration

typedef __bf16 bf16_t;
typedef __attribute__((ext_vector_type(16))) __bf16 v16bf;
typedef __attribute__((ext_vector_type(8)))  __bf16 v8bf;
typedef __attribute__((ext_vector_type(8)))  float  v8f;

union V16 { v16bf v; v8bf h[2]; };

static __device__ __forceinline__ bf16_t f2bf(float f) {
  union { float f; uint32_t u; } x; x.f = f;
  uint32_t r = x.u + 0x7FFFu + ((x.u >> 16) & 1u);   // round-to-nearest-even
  union { uint16_t s; bf16_t b; } y; y.s = (uint16_t)(r >> 16);
  return y.b;
}

// -------------------- stage 1a: f32 -> bf16 flat convert (weights) ---------
__global__ void cvt_f32_bf16(const float* __restrict__ src,
                             bf16_t* __restrict__ dst, int n) {
  int i = blockIdx.x * blockDim.x + threadIdx.x;
  int stride = gridDim.x * blockDim.x;
  for (; i < n; i += stride) dst[i] = f2bf(src[i]);
}

// -------------------- stage 1b: query [B][C][N] -> bf16 n-major [B][N][C] --
__global__ void q_to_nmajor(const float* __restrict__ q,
                            bf16_t* __restrict__ out) {
  int bn = blockIdx.x;                 // b*N + n
  int b = bn >> 12, n = bn & (NN - 1);
  int c = threadIdx.x;                 // 256 threads = CQ
  float v = q[((size_t)b * CQn + c) * NN + n];
  out[(size_t)bn * CQn + c] = f2bf(v);
}

// -------------------- stage 2: bilinear resize kv -> bf16 n-major ----------
__global__ void resize_kv(const float* __restrict__ kv,
                          bf16_t* __restrict__ out) {
  int bn = blockIdx.x;                 // b*N + n
  int b = bn >> 12, n = bn & (NN - 1);
  int oy = n >> 6, ox = n & 63;
  float sy = fmaxf(0.5f * oy - 0.25f, 0.f);  // half-pixel, clamped
  float sx = fmaxf(0.5f * ox - 0.25f, 0.f);
  int y0 = min((int)sy, HKV - 1), x0 = min((int)sx, WKV - 1);
  int y1 = min(y0 + 1, HKV - 1),  x1 = min(x0 + 1, WKV - 1);
  float wy = sy - y0, wx = sx - x0;
  for (int c = threadIdx.x; c < CKVn; c += blockDim.x) {
    const float* p = kv + ((size_t)b * CKVn + c) * (HKV * WKV);
    float v00 = p[y0 * WKV + x0], v01 = p[y0 * WKV + x1];
    float v10 = p[y1 * WKV + x0], v11 = p[y1 * WKV + x1];
    float v = (1.f - wy) * ((1.f - wx) * v00 + wx * v01)
            +        wy  * ((1.f - wx) * v10 + wx * v11);
    out[(size_t)bn * CKVn + c] = f2bf(v);
  }
}

// -------------------- stage 3: WMMA 1x1-conv projection --------------------
// Out[d][n] = sum_c W[d][c] * X[n][c] + bias[d]
// A = W rows (lane l16 row d, K = ck+hi*16 .. +15 contiguous, same order as B)
// B = X n-major: lane l16 column n, K rows ck+hi*16 .. +15 -> one 32B load
// outMode 0: Out[n][D] (for Q/K, m-major)   outMode 1: Out[D][n] (for V)
__global__ void __launch_bounds__(128)
proj_wmma(const bf16_t* __restrict__ X,   // [B][N][C]
          const bf16_t* __restrict__ Wb,  // [D][C]
          const float*  __restrict__ bias,// [D]
          bf16_t* __restrict__ Out,
          int C, int D, int outMode) {
  const int lane = threadIdx.x & 31;
  const int wave = threadIdx.x >> 5;
  const int l16  = lane & 15;
  const int hi   = lane >> 4;
  const int ntiles = NN / 16, dtiles = D / 16;
  int wg  = blockIdx.x * 4 + wave;
  int tpb = dtiles * ntiles;
  int b   = wg / tpb;
  if (b >= BB) return;
  int rem = wg - b * tpb;
  int dt  = rem / ntiles;
  int nt  = rem - dt * ntiles;

  const bf16_t* xrow = X + ((size_t)b * NN + (nt * 16 + l16)) * C + hi * 16;
  const bf16_t* wrow = Wb + (size_t)(dt * 16 + l16) * C + hi * 16;

  v8f acc = {};
  for (int ck = 0; ck < C; ck += 32) {
    V16 a, bm;
    a.h[0] = *(const v8bf*)(wrow + ck);
    a.h[1] = *(const v8bf*)(wrow + ck + 8);
    bm.v   = *(const v16bf*)(xrow + ck);
    acc = __builtin_amdgcn_wmma_f32_16x16x32_bf16(
        false, a.v, false, bm.v, (short)0, acc, false, false);
  }
  int n = nt * 16 + l16;
#pragma unroll
  for (int j = 0; j < 8; ++j) {             // C-layout: slot j -> row d=j+8*hi
    int d = dt * 16 + j + 8 * hi;
    float v = acc[j] + bias[d];
    if (outMode == 0) Out[((size_t)b * NN + n) * D + d] = f2bf(v);
    else              Out[((size_t)b * D + d) * NN + n] = f2bf(v);
  }
}

// -------------------- stage 4: fused flash attention -----------------------
// One wave == 16 query rows; one block == 4 waves == 64 rows of one batch.
// K/V m-chunks are staged cooperatively into LDS with double-buffered
// global_load_async_to_lds_b128 (9 per wave per chunk), overlapped with the
// WMMA consumption of the previous chunk; s_wait_asynccnt 9 retires exactly
// the in-order older chunk. Per chunk: 2 QK^T WMMAs + online softmax
// (half-wave shfl reductions) + LDS P transpose + 16 PV WMMAs into a
// 16x256 f32 accumulator held in 128 VGPRs.
__global__ void __launch_bounds__(128, 1)
flash_attn(const bf16_t* __restrict__ Qm,   // [B][N][32] n-major
           const bf16_t* __restrict__ Km,   // [B][N][32] m-major
           const bf16_t* __restrict__ Vm,   // [B][CQ][N] c-major
           const float*  __restrict__ query,// [B][CQ][N]
           const float*  __restrict__ gamma,
           float* __restrict__ out) {
  __shared__ __attribute__((aligned(32))) bf16_t Kbuf[2][MCH * DQKn];  // 2x2KB
  __shared__ __attribute__((aligned(32))) bf16_t Vbuf[2][CQn * MCH];   // 2x16KB
  __shared__ __attribute__((aligned(32))) bf16_t ldsP[4][16][32];      // 4KB

  const int tid  = threadIdx.x;
  const int lane = tid & 31;
  const int wave = tid >> 5;
  const int l16  = lane & 15;
  const int hi   = lane >> 4;

  int wg    = blockIdx.x * 4 + wave;  // 0..2047
  int b     = wg >> 8;                // 256 n-tiles per batch
  int nbase = (wg & 255) * 16;

  const bf16_t* Kb = Km + (size_t)b * NN * DQKn;
  const bf16_t* Vb = Vm + (size_t)b * CQn * NN;

  // Q A-tile (row n = nbase+l16; K = hi*16..+15 contiguous)
  const bf16_t* qrow = Qm + ((size_t)b * NN + (nbase + l16)) * DQKn;
  V16 qa;
  qa.h[0] = *(const v8bf*)(qrow + hi * 16);
  qa.h[1] = *(const v8bf*)(qrow + hi * 16 + 8);

  // async stage of one m-chunk into LDS buffer `buf` (9 x b128 per wave)
  auto stage = [&](int mb, int buf) {
    uint32_t koff = (uint32_t)(size_t)&Kbuf[buf][0] + (uint32_t)tid * 16u;
    unsigned long long kg =
        (unsigned long long)(size_t)(Kb + (size_t)mb * DQKn) + (size_t)tid * 16u;
    asm volatile("global_load_async_to_lds_b128 %0, %1, off"
                 :: "v"(koff), "v"(kg) : "memory");
    int c0 = tid >> 2, q = tid & 3;
#pragma unroll
    for (int r = 0; r < 8; ++r) {
      int c = r * 32 + c0;
      uint32_t voff = (uint32_t)(size_t)&Vbuf[buf][0] + (uint32_t)(c * 64 + q * 16);
      unsigned long long vg =
          (unsigned long long)(size_t)(Vb + (size_t)c * NN + mb) + (size_t)q * 16u;
      asm volatile("global_load_async_to_lds_b128 %0, %1, off"
                   :: "v"(voff), "v"(vg) : "memory");
    }
  };

  v8f zero = {};
  v8f acc[16];
#pragma unroll
  for (int t = 0; t < 16; ++t) acc[t] = zero;
  float mrow[8], lrow[8];
#pragma unroll
  for (int j = 0; j < 8; ++j) { mrow[j] = -__builtin_inff(); lrow[j] = 0.f; }
  const float g = gamma[0];

  stage(0, 0);                                   // prologue: chunk 0 in flight

  for (int it = 0; it < NN / MCH; ++it) {        // 128 iterations
    const int mb  = it * MCH;
    const int cur = it & 1;
    if (it + 1 < NN / MCH) {
      stage(mb + MCH, cur ^ 1);                  // overlap next chunk
      asm volatile("s_wait_asynccnt 0x9" ::: "memory");  // oldest 9 (cur) done
    } else {
      asm volatile("s_wait_asynccnt 0x0" ::: "memory");
    }
    __syncthreads();                             // all waves' stages visible

    const bf16_t* kcur = &Kbuf[cur][0];
    const bf16_t* vcur = &Vbuf[cur][0];

    // E tiles: column m = {0,16}+l16 of the chunk, K rows d = hi*16..+15
    V16 kb0, kb1;
    kb0.v = *(const v16bf*)(kcur + (size_t)l16 * DQKn + hi * 16);
    kb1.v = *(const v16bf*)(kcur + (size_t)(16 + l16) * DQKn + hi * 16);
    v8f e0 = __builtin_amdgcn_wmma_f32_16x16x32_bf16(
        false, qa.v, false, kb0.v, (short)0, zero, false, false);
    v8f e1 = __builtin_amdgcn_wmma_f32_16x16x32_bf16(
        false, qa.v, false, kb1.v, (short)0, zero, false, false);

    float p0[8], p1[8], scale[8];
#pragma unroll
    for (int j = 0; j < 8; ++j) {
      float tm = fmaxf(e0[j], e1[j]);          // row M=j+8*hi, cols over lanes
      tm = fmaxf(tm, __shfl_xor(tm, 1, 32));
      tm = fmaxf(tm, __shfl_xor(tm, 2, 32));
      tm = fmaxf(tm, __shfl_xor(tm, 4, 32));
      tm = fmaxf(tm, __shfl_xor(tm, 8, 32));   // stays within 16-lane half
      float nm = fmaxf(mrow[j], tm);
      scale[j] = __expf(mrow[j] - nm);
      mrow[j]  = nm;
      p0[j] = __expf(e0[j] - nm);
      p1[j] = __expf(e1[j] - nm);
      float s = p0[j] + p1[j];
      s += __shfl_xor(s, 1, 32);
      s += __shfl_xor(s, 2, 32);
      s += __shfl_xor(s, 4, 32);
      s += __shfl_xor(s, 8, 32);
      lrow[j] = lrow[j] * scale[j] + s;
    }
#pragma unroll
    for (int t = 0; t < 16; ++t)
#pragma unroll
      for (int j = 0; j < 8; ++j)
        acc[t][j] *= scale[j];                 // slot-aligned row rescale

    // P: C-layout -> A-layout through wave-private LDS (in-order pipeline)
#pragma unroll
    for (int j = 0; j < 8; ++j) {
      ldsP[wave][j + 8 * hi][l16]      = f2bf(p0[j]);
      ldsP[wave][j + 8 * hi][16 + l16] = f2bf(p1[j]);
    }
    asm volatile("s_wait_dscnt 0" ::: "memory");
    V16 pa;
    pa.h[0] = *(const v8bf*)(&ldsP[wave][l16][hi * 16]);
    pa.h[1] = *(const v8bf*)(&ldsP[wave][l16][hi * 16 + 8]);
    asm volatile("" ::: "memory");

    // acc[t](16n x 16c) += P(16n x 32m) x V(32m x 16c), V from shared LDS
#pragma unroll
    for (int t = 0; t < 16; ++t) {
      V16 vb;
      vb.v = *(const v16bf*)(vcur + (size_t)(t * 16 + l16) * MCH + hi * 16);
      acc[t] = __builtin_amdgcn_wmma_f32_16x16x32_bf16(
          false, pa.v, false, vb.v, (short)0, acc[t], false, false);
    }
    __syncthreads();                           // cur buffer free for overwrite
  }

  // epilogue: out = gamma * (acc / l) + query
  float rinv[8];
#pragma unroll
  for (int j = 0; j < 8; ++j) rinv[j] = 1.f / lrow[j];
#pragma unroll
  for (int t = 0; t < 16; ++t) {
    int c = t * 16 + l16;
#pragma unroll
    for (int j = 0; j < 8; ++j) {
      int n = nbase + j + 8 * hi;
      size_t idx = ((size_t)b * CQn + c) * NN + n;
      out[idx] = g * (acc[t][j] * rinv[j]) + query[idx];
    }
  }
}

// ---------------------------------------------------------------------------
extern "C" void kernel_launch(void* const* d_in, const int* in_sizes, int n_in,
                              void* d_out, int out_size, void* d_ws, size_t ws_size,
                              hipStream_t stream) {
  const float* query     = (const float*)d_in[0];
  const float* key_value = (const float*)d_in[1];
  const float* Wq        = (const float*)d_in[2];
  const float* bq        = (const float*)d_in[3];
  const float* Wk        = (const float*)d_in[4];
  const float* bk        = (const float*)d_in[5];
  const float* Wv        = (const float*)d_in[6];
  const float* bv        = (const float*)d_in[7];
  const float* gamma     = (const float*)d_in[8];
  float* out = (float*)d_out;

  char* ws = (char*)d_ws;
  size_t off = 0;
  bf16_t* qbf = (bf16_t*)(ws + off); off += (size_t)BB * NN * CQn  * 2; // 16.8MB
  bf16_t* kvr = (bf16_t*)(ws + off); off += (size_t)BB * NN * CKVn * 2; // 33.6MB
  bf16_t* Wqb = (bf16_t*)(ws + off); off += (size_t)DQKn * CQn  * 2;
  bf16_t* Wkb = (bf16_t*)(ws + off); off += (size_t)DQKn * CKVn * 2;
  bf16_t* Wvb = (bf16_t*)(ws + off); off += (size_t)CQn  * CKVn * 2;
  bf16_t* Qm  = (bf16_t*)(ws + off); off += (size_t)BB * NN * DQKn * 2; // 2.1MB
  bf16_t* Km  = (bf16_t*)(ws + off); off += (size_t)BB * NN * DQKn * 2; // 2.1MB
  bf16_t* Vm  = (bf16_t*)(ws + off); off += (size_t)BB * CQn * NN  * 2; // 16.8MB

  // stage 1: precision conversion / layout
  cvt_f32_bf16<<<32,  256, 0, stream>>>(Wq, Wqb, DQKn * CQn);
  cvt_f32_bf16<<<64,  256, 0, stream>>>(Wk, Wkb, DQKn * CKVn);
  cvt_f32_bf16<<<512, 256, 0, stream>>>(Wv, Wvb, CQn * CKVn);
  q_to_nmajor<<<BB * NN, 256, 0, stream>>>(query, qbf);

  // stage 2: bilinear resize
  resize_kv<<<BB * NN, 256, 0, stream>>>(key_value, kvr);

  // stage 3: WMMA projections (Q,K m-major [N][32]; V c-major [256][N])
  proj_wmma<<<(BB * 2  * 256) / 4, 128, 0, stream>>>(qbf, Wqb, bq, Qm, CQn,  DQKn, 0);
  proj_wmma<<<(BB * 2  * 256) / 4, 128, 0, stream>>>(kvr, Wkb, bk, Km, CKVn, DQKn, 0);
  proj_wmma<<<(BB * 16 * 256) / 4, 128, 0, stream>>>(kvr, Wvb, bv, Vm, CKVn, CQn,  1);

  // stage 4: fused flash attention + residual
  flash_attn<<<(BB * 256) / 4, 128, 0, stream>>>(Qm, Km, Vm, query, gamma, out);
}
